// MambaAttWrapper_49615462204122
// MI455X (gfx1250) — matrix operand order repaired and checked by reading, not compiled, not run on hardware
//
#include <hip/hip_runtime.h>
#include <hip/hip_bf16.h>
#include <math.h>

#define D_MODEL   1024
#define HEADDIM   64
#define D_STATE   128
#define D_CONV    4
#define D_INNER   2048
#define NHEADS    32
#define CONV_DIM  2304          // D_INNER + 2*D_STATE
#define D_IN_PROJ 4384          // 2*D_INNER + 2*D_STATE + NHEADS
#define D_IN_PROJ_PAD 4416      // next multiple of 64 (69 tiles)
#define BATCH     4
#define SEQ       2048
#define BL        (BATCH*SEQ)   // 8192 rows

typedef __attribute__((ext_vector_type(16))) __bf16 bf16x16;
typedef __attribute__((ext_vector_type(8)))  float  f32x8;

union FragAB { bf16x16 v; uint4 q[2]; unsigned short h[16]; };
union FragC  { f32x8   v; float f[8]; };

__device__ __forceinline__ unsigned short f2bf(float x) {
  unsigned u = __float_as_uint(x);
  u += 0x7FFFu + ((u >> 16) & 1u);           // round-to-nearest-even
  return (unsigned short)(u >> 16);
}

// ------------------------------------------- f32 -> bf16, zero-padded, x4 wide
__global__ void mamba_cvt_bf16(const float* __restrict__ src,
                               unsigned short* __restrict__ dst,
                               int n_src, int n_dst) {
  int i4 = (blockIdx.x * blockDim.x + threadIdx.x) * 4;
  if (i4 >= n_dst) return;
  ushort4 o;
  o.x = (i4 + 0 < n_src) ? f2bf(src[i4 + 0]) : (unsigned short)0;
  o.y = (i4 + 1 < n_src) ? f2bf(src[i4 + 1]) : (unsigned short)0;
  o.z = (i4 + 2 < n_src) ? f2bf(src[i4 + 2]) : (unsigned short)0;
  o.w = (i4 + 3 < n_src) ? f2bf(src[i4 + 3]) : (unsigned short)0;
  *(ushort4*)(dst + i4) = o;
}

// --------------------------------------------------- WMMA bf16 GEMM, C = A*B^T
// A: MxK bf16 row-major (M % 64 == 0), Bw: N_pad x K bf16 row-major
// (N_pad = gridDim.x*64, rows >= N zero-padded), C: MxN f32 row-major.
// One wave per 64x64 tile, 4x4 grid of v_wmma_f32_16x16x32_bf16, guard-free
// K-loop, branchless near-scope prefetch of the next K tile.
__global__ __launch_bounds__(32)
void mamba_gemm_bf16_wmma(const unsigned short* __restrict__ A,
                          const unsigned short* __restrict__ Bw,
                          float* __restrict__ C,
                          int M, int N, int K) {
  const int lane = threadIdx.x;
  const int lnm  = lane & 15;
  const int hi   = lane >> 4;         // 0 or 1 (lane half)
  const int m0   = blockIdx.y * 64;
  const int n0   = blockIdx.x * 64;

  FragC acc[4][4];
#pragma unroll
  for (int mi = 0; mi < 4; ++mi)
#pragma unroll
    for (int ni = 0; ni < 4; ++ni)
#pragma unroll
      for (int j = 0; j < 8; ++j) acc[mi][ni].f[j] = 0.0f;

  const int selA = hi * 8;            // A: lanes16-31 hold K base +8 / +24
  const int selB = hi * 8;            // B: lanes16-31 hold K base +8 / +24

  const unsigned short* pa[4];
  const unsigned short* pb[4];
#pragma unroll
  for (int mi = 0; mi < 4; ++mi)
    pa[mi] = A + (size_t)(m0 + mi * 16 + lnm) * K + selA;
#pragma unroll
  for (int ni = 0; ni < 4; ++ni)
    pb[ni] = Bw + (size_t)(n0 + ni * 16 + lnm) * K + selB;

  for (int k0 = 0; k0 < K; k0 += 32) {
    const int pf = (k0 + 32 < K) ? (k0 + 32) : k0;   // branchless prefetch clamp
    FragAB a[4], b[4];
#pragma unroll
    for (int mi = 0; mi < 4; ++mi) {
      a[mi].q[0] = *(const uint4*)(pa[mi] + k0);        // K +0..7
      a[mi].q[1] = *(const uint4*)(pa[mi] + k0 + 16);   // K +16..23
    }
    __builtin_prefetch(pa[0] + pf, 0, 3);
#pragma unroll
    for (int ni = 0; ni < 4; ++ni) {
      b[ni].q[0] = *(const uint4*)(pb[ni] + k0);        // K +0..7
      b[ni].q[1] = *(const uint4*)(pb[ni] + k0 + 16);   // K +16..23
    }
    __builtin_prefetch(pb[0] + pf, 0, 3);
#pragma unroll
    for (int mi = 0; mi < 4; ++mi)
#pragma unroll
      for (int ni = 0; ni < 4; ++ni)
        acc[mi][ni].v = __builtin_amdgcn_wmma_f32_16x16x32_bf16(
            false, a[mi].v, false, b[ni].v,
            (short)0, acc[mi][ni].v, false, false);
  }

  // C/D layout: vector elem j -> M = j (lanes 0-15) / j+8 (lanes 16-31), N = lane%16
#pragma unroll
  for (int mi = 0; mi < 4; ++mi)
#pragma unroll
    for (int ni = 0; ni < 4; ++ni) {
      const int col = n0 + ni * 16 + lnm;
      if (col < N) {
#pragma unroll
        for (int j = 0; j < 8; ++j) {
          const int row = m0 + mi * 16 + j + hi * 8;
          C[(size_t)row * N + col] = acc[mi][ni].f[j];
        }
      }
    }
}

// ------------------------------------------------- causal depthwise conv + SiLU
__global__ void mamba_conv_silu(const float* __restrict__ zx,
                                const float* __restrict__ cw,
                                const float* __restrict__ cb,
                                float* __restrict__ out) {
  int idx = blockIdx.x * blockDim.x + threadIdx.x;   // BL*CONV_DIM
  if (idx >= BL * CONV_DIM) return;
  const int c  = idx % CONV_DIM;
  const int bl = idx / CONV_DIM;
  const int l  = bl % SEQ;
  float acc = cb[c];
  const float* base = zx + (size_t)bl * D_IN_PROJ + D_INNER + c;
#pragma unroll
  for (int k = 0; k < D_CONV; ++k) {
    const int dl = l + k - (D_CONV - 1);
    if (dl >= 0)
      acc += base[(long)(k - (D_CONV - 1)) * D_IN_PROJ] * cw[c * D_CONV + k];
  }
  out[idx] = acc / (1.0f + expf(-acc));              // SiLU
}

// ----------------------------------------------------- dt = softplus, dA = e^{dtA}
__global__ void mamba_dt(const float* __restrict__ zx,
                         const float* __restrict__ dt_bias,
                         const float* __restrict__ A_log,
                         float* __restrict__ dt_out,
                         float* __restrict__ dA_out) {
  int idx = blockIdx.x * blockDim.x + threadIdx.x;   // BL*NHEADS
  if (idx >= BL * NHEADS) return;
  const int h  = idx % NHEADS;
  const int bl = idx / NHEADS;
  const float raw = zx[(size_t)bl * D_IN_PROJ + (D_IN_PROJ - NHEADS) + h]
                    + dt_bias[h];
  const float dt = raw > 20.0f ? raw : log1pf(expf(raw));
  dt_out[idx] = dt;
  dA_out[idx] = expf(-dt * expf(A_log[h]));
}

// ------------------------------------------------------------- selective scan
// One block per (batch, head). State h[64][128] register-resident:
// thread owns p = tid&63, n = (tid>>6)*32 .. +31 (32 floats).
__global__ __launch_bounds__(256)
void mamba_scan(const float* __restrict__ xBCa,
                const float* __restrict__ dtA,
                const float* __restrict__ dAA,
                const float* __restrict__ Dp,
                float* __restrict__ Y) {
  __shared__ float stage[320];     // x[64] | B[128] | C[128]
  __shared__ float partial[256];
  const int bh = blockIdx.x;
  const int b  = bh / NHEADS;
  const int h  = bh % NHEADS;
  const int tid = threadIdx.x;
  const int p  = tid & 63;
  const int nb = tid >> 6;
  const float Dh = Dp[h];
  const float* xs = stage;
  const float* Bn = stage + 64  + nb * 32;
  const float* Cn = stage + 192 + nb * 32;

  float hst[32];
#pragma unroll
  for (int j = 0; j < 32; ++j) hst[j] = 0.0f;

  for (int t = 0; t < SEQ; ++t) {
    const size_t row = (size_t)(b * SEQ + t);
    const size_t rbase = row * CONV_DIM;
    // stage x (cols h*64..), then B|C (cols D_INNER..D_INNER+255, contiguous)
    for (int i = tid; i < 320; i += 256)
      stage[i] = xBCa[rbase + (i < 64 ? h * HEADDIM + i : D_INNER + (i - 64))];
    const float dA_t = dAA[row * NHEADS + h];
    const float dt_t = dtA[row * NHEADS + h];
    __syncthreads();

    const float dtx = dt_t * xs[p];
    float acc = 0.0f;
#pragma unroll
    for (int j = 0; j < 32; ++j) {
      hst[j] = hst[j] * dA_t + dtx * Bn[j];
      acc += hst[j] * Cn[j];
    }
    partial[nb * 64 + p] = acc;
    __syncthreads();

    if (tid < 64) {
      const float y = partial[p] + partial[64 + p] + partial[128 + p] +
                      partial[192 + p] + Dh * xs[p];
      Y[row * D_INNER + h * HEADDIM + p] = y;
    }
    __syncthreads();   // protect stage/partial reuse
  }
}

// ------------------------------------------- y*silu(z), RMSNorm, emit bf16 rows
__global__ __launch_bounds__(256)
void mamba_gate_norm(const float* __restrict__ Y,
                     const float* __restrict__ zx,
                     const float* __restrict__ nw,
                     unsigned short* __restrict__ ybf) {
  const int bl  = blockIdx.x;
  const int tid = threadIdx.x;
  __shared__ float red[256];
  const float* yrow = Y  + (size_t)bl * D_INNER;
  const float* zrow = zx + (size_t)bl * D_IN_PROJ;
  float g[8];
  float ss = 0.0f;
#pragma unroll
  for (int j = 0; j < 8; ++j) {
    const int c = tid + j * 256;
    const float z = zrow[c];
    const float v = yrow[c] * (z / (1.0f + expf(-z)));
    g[j] = v;
    ss += v * v;
  }
  red[tid] = ss;
  __syncthreads();
  for (int s = 128; s > 0; s >>= 1) {
    if (tid < s) red[tid] += red[tid + s];
    __syncthreads();
  }
  const float scale = rsqrtf(red[0] / (float)D_INNER + 1e-5f);
#pragma unroll
  for (int j = 0; j < 8; ++j) {
    const int c = tid + j * 256;
    ybf[(size_t)bl * D_INNER + c] = f2bf(g[j] * scale * nw[c]);
  }
}

// ============================================================================
extern "C" void kernel_launch(void* const* d_in, const int* in_sizes, int n_in,
                              void* d_out, int out_size, void* d_ws, size_t ws_size,
                              hipStream_t stream) {
  const float* u          = (const float*)d_in[0];   // query
  const float* in_proj_w  = (const float*)d_in[3];
  const float* conv_w     = (const float*)d_in[4];
  const float* conv_b     = (const float*)d_in[5];
  const float* dt_bias    = (const float*)d_in[6];
  const float* A_log      = (const float*)d_in[7];
  const float* Dp         = (const float*)d_in[8];
  const float* norm_w     = (const float*)d_in[9];
  const float* out_proj_w = (const float*)d_in[10];
  float* out = (float*)d_out;

  char* ws = (char*)d_ws;
  size_t off = 0;
  auto alloc = [&](size_t bytes) -> void* {
    void* p = ws + off;
    off += (bytes + 255) & ~(size_t)255;
    return p;
  };
  unsigned short* u_bf  = (unsigned short*)alloc((size_t)BL * D_MODEL * 2);
  unsigned short* w1_bf = (unsigned short*)alloc((size_t)D_IN_PROJ_PAD * D_MODEL * 2);
  unsigned short* wo_bf = (unsigned short*)alloc((size_t)D_MODEL * D_INNER * 2);
  float* zx   = (float*)alloc((size_t)BL * D_IN_PROJ * 4);
  float* xBCa = (float*)alloc((size_t)BL * CONV_DIM * 4);
  float* dtA  = (float*)alloc((size_t)BL * NHEADS * 4);
  float* dAA  = (float*)alloc((size_t)BL * NHEADS * 4);
  float* Yb   = (float*)alloc((size_t)BL * D_INNER * 4);
  unsigned short* y_bf = (unsigned short*)alloc((size_t)BL * D_INNER * 2);
  (void)ws_size; (void)in_sizes; (void)n_in; (void)out_size;

  // 1) down-convert activations + weights to bf16 (zero-pad in_proj_w rows)
  {
    int n1 = BL * D_MODEL;
    mamba_cvt_bf16<<<(n1 / 4 + 255) / 256, 256, 0, stream>>>(u, u_bf, n1, n1);
    int n2s = D_IN_PROJ * D_MODEL, n2d = D_IN_PROJ_PAD * D_MODEL;
    mamba_cvt_bf16<<<(n2d / 4 + 255) / 256, 256, 0, stream>>>(in_proj_w, w1_bf,
                                                              n2s, n2d);
    int n3 = D_MODEL * D_INNER;
    mamba_cvt_bf16<<<(n3 / 4 + 255) / 256, 256, 0, stream>>>(out_proj_w, wo_bf,
                                                             n3, n3);
  }

  // 2) in_proj GEMM: zx[8192 x 4384] = u[8192 x 1024] * W^T  (guard-free K loop)
  {
    dim3 grid(D_IN_PROJ_PAD / 64, BL / 64);
    mamba_gemm_bf16_wmma<<<grid, 32, 0, stream>>>(u_bf, w1_bf, zx,
                                                  BL, D_IN_PROJ, D_MODEL);
  }

  // 3) causal conv + SiLU over xBC columns
  {
    int n = BL * CONV_DIM;
    mamba_conv_silu<<<(n + 255) / 256, 256, 0, stream>>>(zx, conv_w, conv_b, xBCa);
  }

  // 4) dt = softplus(raw + bias), dA = exp(dt * A)
  {
    int n = BL * NHEADS;
    mamba_dt<<<(n + 255) / 256, 256, 0, stream>>>(zx, dt_bias, A_log, dtA, dAA);
  }

  // 5) selective scan: 128 blocks (B x NHEADS), register-resident state
  mamba_scan<<<BATCH * NHEADS, 256, 0, stream>>>(xBCa, dtA, dAA, Dp, Yb);

  // 6) gate + RMSNorm, emit bf16
  mamba_gate_norm<<<BL, 256, 0, stream>>>(Yb, zx, norm_w, y_bf);

  // 7) out_proj GEMM: out[8192 x 1024] = y[8192 x 2048] * Wout^T
  {
    dim3 grid(D_MODEL / 64, BL / 64);
    mamba_gemm_bf16_wmma<<<grid, 32, 0, stream>>>(y_bf, wo_bf, out,
                                                  BL, D_MODEL, D_INNER);
  }
}
